// AgentAttention_85194971283973
// MI455X (gfx1250) — matrix operand order repaired and checked by reading, not compiled, not run on hardware
//
#include <hip/hip_runtime.h>
#include <hip/hip_bf16.h>

// ---------------- problem constants ----------------
#define Bb   16
#define Nn   3136
#define Cc   512
#define Hh   8
#define Dd   64
#define Ag   49
#define W56  56

typedef __bf16 bf16;
typedef __attribute__((ext_vector_type(16))) __bf16 v16bf;
typedef __attribute__((ext_vector_type(8)))  __bf16 v8bf;
typedef __attribute__((ext_vector_type(8)))  float  v8f;

// ---------------- WMMA helpers (CDNA5 16x16x32 bf16 layouts) ----------------
// A (16x32, row-major source): lane&15 = M row; lane<16 -> K chunks [0..7],[16..23];
// lane>=16 -> [8..15],[24..31].
__device__ __forceinline__ v16bf wmma_load_a(const bf16* rowptr, int lane) {
    int khalf = (lane & 16) ? 8 : 0;
    union { v16bf v; v8bf h[2]; } u;
    u.h[0] = *(const v8bf*)(rowptr + khalf);
    u.h[1] = *(const v8bf*)(rowptr + khalf + 16);
    return u.v;
}
// B = Bt^T where Bt is (N x K) row-major; lane&15 = N col; lane<16 -> K chunks [0..7],[16..23];
// lane>=16 -> [8..15],[24..31].
__device__ __forceinline__ v16bf wmma_load_bt(const bf16* rowptr, int lane) {
    int ksec = (lane & 16) ? 8 : 0;
    union { v16bf v; v8bf h[2]; } u;
    u.h[0] = *(const v8bf*)(rowptr + ksec);
    u.h[1] = *(const v8bf*)(rowptr + ksec + 16);
    return u.v;
}
__device__ __forceinline__ v8f wmma_bf16(v16bf a, v16bf b, v8f c) {
    return __builtin_amdgcn_wmma_f32_16x16x32_bf16(false, a, false, b, (short)0, c,
                                                   false, false);
}

// ---------------- elementwise: f32 -> bf16, 8 elems/thread ----------------
__global__ void k_cast8(const float* __restrict__ src, bf16* __restrict__ dst, int n8) {
    int i = blockIdx.x * blockDim.x + threadIdx.x;
    if (i >= n8) return;
    const float4* s = (const float4*)src + (size_t)i * 2;
    float4 f0 = s[0], f1 = s[1];
    v8bf o;
    o[0] = (bf16)f0.x; o[1] = (bf16)f0.y; o[2] = (bf16)f0.z; o[3] = (bf16)f0.w;
    o[4] = (bf16)f1.x; o[5] = (bf16)f1.y; o[6] = (bf16)f1.z; o[7] = (bf16)f1.w;
    ((v8bf*)dst)[i] = o;
}

// ---------------- QKV GEMM: (50176 x 512) * (1536 x 512)^T, 2x2 reg tiles ------
__global__ __launch_bounds__(256) void k_qkv(const bf16* __restrict__ xb,
                                             const bf16* __restrict__ wb,
                                             bf16* __restrict__ qb,
                                             bf16* __restrict__ kb,
                                             bf16* __restrict__ vb) {
    int lane = threadIdx.x & 31;
    int tile = blockIdx.x * (blockDim.x >> 5) + (threadIdx.x >> 5);
    const int TN2 = (3 * Cc) / 32; // 48
    int mt = (tile / TN2) * 32;
    int jt = (tile % TN2) * 32;
    v8f a00 = {}, a01 = {}, a10 = {}, a11 = {};
    const bf16* arow0 = xb + (size_t)(mt + (lane & 15)) * Cc;
    const bf16* arow1 = arow0 + (size_t)16 * Cc;
    const bf16* brow0 = wb + (size_t)(jt + (lane & 15)) * Cc;
    const bf16* brow1 = brow0 + (size_t)16 * Cc;
#pragma unroll 4
    for (int k0 = 0; k0 < Cc; k0 += 32) {
        __builtin_prefetch(arow0 + k0 + 128, 0, 1);
        v16bf a0 = wmma_load_a(arow0 + k0, lane);
        v16bf a1 = wmma_load_a(arow1 + k0, lane);
        v16bf b0 = wmma_load_bt(brow0 + k0, lane);
        v16bf b1 = wmma_load_bt(brow1 + k0, lane);
        a00 = wmma_bf16(a0, b0, a00);
        a01 = wmma_bf16(a0, b1, a01);
        a10 = wmma_bf16(a1, b0, a10);
        a11 = wmma_bf16(a1, b1, a11);
    }
    int hi8 = (lane & 16) ? 8 : 0;
#pragma unroll
    for (int mo = 0; mo < 2; ++mo) {
#pragma unroll
        for (int jo = 0; jo < 2; ++jo) {
            v8f acc = mo ? (jo ? a11 : a10) : (jo ? a01 : a00);
            int j = jt + jo * 16 + (lane & 15);
            bf16* dst; int jj;
            if (j < Cc)          { dst = qb; jj = j; }
            else if (j < 2 * Cc) { dst = kb; jj = j - Cc; }
            else                 { dst = vb; jj = j - 2 * Cc; }
            int mrow = mt + mo * 16 + hi8;
#pragma unroll
            for (int r = 0; r < 8; ++r) dst[(size_t)(mrow + r) * Cc + jj] = (bf16)acc[r];
        }
    }
}

// ------- V transpose via async global->LDS copy: vb (b,n,c) -> vtb (b,c,n) -----
// Tile 32(n) x 32(c). Each thread async-copies 8B (4 bf16) straight into LDS
// (ASYNCcnt-tracked, no VGPR round trip), then writes transposed, coalesced in n.
__global__ void k_vt(const bf16* __restrict__ vb, bf16* __restrict__ vtb) {
    __shared__ bf16 t[32][36]; // row stride 72B: 8B-aligned, conflict-free col reads
    int bt = blockIdx.x;                 // b*(16*98) + ct*98 + ntb
    int b  = bt / (16 * 98);
    int r  = bt % (16 * 98);
    int ct = r / 98, ntb = r % 98;
    int c0 = ct * 32, n0 = ntb * 32;
    int tid = threadIdx.y * 32 + threadIdx.x;  // 0..255
    int row = tid >> 3;                        // n-local 0..31
    int seg = tid & 7;                         // 8B segment within row
    const bf16* gsrc = vb + ((size_t)b * Nn + n0 + row) * Cc + c0 + seg * 4;
    unsigned lds_addr = (unsigned)(unsigned long long)&t[row][seg * 4];
    unsigned long long gaddr = (unsigned long long)gsrc;
    asm volatile("global_load_async_to_lds_b64 %0, %1, off"
                 :: "v"(lds_addr), "v"(gaddr) : "memory");
    asm volatile("s_wait_asynccnt 0" ::: "memory");
    __syncthreads();
    int tx = threadIdx.x, ty = threadIdx.y;
#pragma unroll
    for (int i = 0; i < 4; ++i) {
        int c = c0 + ty + 8 * i;
        vtb[((size_t)b * Cc + c) * Nn + n0 + tx] = t[tx][ty + 8 * i];
    }
}

// ---------------- agent pooling: 8x8 block mean of q, rows padded to 64 --------
__global__ void k_pool(const bf16* __restrict__ qb, bf16* __restrict__ agentb) {
    int idx = blockIdx.x * blockDim.x + threadIdx.x; // ((b*64)+ar)*Cc + c
    if (idx >= Bb * 64 * Cc) return;
    int c  = idx % Cc;
    int ar = (idx / Cc) & 63;
    int b  = idx / (Cc * 64);
    float s = 0.f;
    if (ar < Ag) {
        int ay = ar / 7, ax = ar % 7;
        for (int yy = 0; yy < 8; ++yy)
            for (int xx = 0; xx < 8; ++xx)
                s += (float)qb[((size_t)(b * Nn) + (ay * 8 + yy) * W56 + ax * 8 + xx) * Cc + c];
        s *= (1.0f / 64.0f);
    }
    agentb[idx] = (bf16)s;
}

// ---------------- bilinear helpers (jax.image.resize upsample semantics) -------
__device__ __forceinline__ void lin_coord(float pos, int n, int& i0, int& i1, float& f) {
    float fl = floorf(pos);
    f  = pos - fl;
    int a = (int)fl, b = a + 1;
    i0 = a < 0 ? 0 : (a > n - 1 ? n - 1 : a);
    i1 = b < 0 ? 0 : (b > n - 1 ? n - 1 : b);
}
__device__ __forceinline__ float resize7to14(const float* p /*7x7*/, int wy, int wx) {
    int y0, y1, x0, x1; float fy, fx;
    lin_coord((wy + 0.5f) * 0.5f - 0.5f, 7, y0, y1, fy);
    lin_coord((wx + 0.5f) * 0.5f - 0.5f, 7, x0, x1, fx);
    float v00 = p[y0 * 7 + x0], v01 = p[y0 * 7 + x1];
    float v10 = p[y1 * 7 + x0], v11 = p[y1 * 7 + x1];
    return (1.f - fy) * ((1.f - fx) * v00 + fx * v01) +
           fy        * ((1.f - fx) * v10 + fx * v11);
}

// bias1[h][a][n]  (stage-1 positional bias, upsampled 196 -> 3136)
__global__ void k_bias1(const float* __restrict__ an, const float* __restrict__ ahb,
                        const float* __restrict__ awb, float* __restrict__ bias1) {
    int idx = blockIdx.x * blockDim.x + threadIdx.x;
    if (idx >= Hh * Ag * Nn) return;
    int n = idx % Nn, a = (idx / Nn) % Ag, h = idx / (Nn * Ag);
    int i0, i1; float f;
    lin_coord((n + 0.5f) * (196.0f / 3136.0f) - 0.5f, 196, i0, i1, f);
    const float* an_p = an  + (h * Ag + a) * 49;
    const float* ah_p = ahb + (h * Ag + a) * 14;
    const float* aw_p = awb + (h * Ag + a) * 14;
    int wy0 = i0 / 14, wx0 = i0 % 14, wy1 = i1 / 14, wx1 = i1 % 14;
    float v0 = resize7to14(an_p, wy0, wx0) + ah_p[wy0] + aw_p[wx0];
    float v1 = resize7to14(an_p, wy1, wx1) + ah_p[wy1] + aw_p[wx1];
    bias1[idx] = (1.f - f) * v0 + f * v1;
}

__device__ __forceinline__ float ab_val(const float* na, const float* hab,
                                        const float* wab, const float* cab,
                                        int h, int r, int a) {
    if (r == 0) return cab[h * Ag + a];
    int w = r - 1, wy = w / 14, wx = w % 14;
    return resize7to14(na + (h * Ag + a) * 49, wy, wx) +
           hab[(h * 14 + wy) * Ag + a] + wab[(h * 14 + wx) * Ag + a];
}
// bias2[h][n][a]  (stage-2 positional bias, rows: [ca | 196 window rows] -> 3136)
__global__ void k_bias2(const float* __restrict__ na, const float* __restrict__ hab,
                        const float* __restrict__ wab, const float* __restrict__ cab,
                        float* __restrict__ bias2) {
    int idx = blockIdx.x * blockDim.x + threadIdx.x;
    if (idx >= Hh * Nn * Ag) return;
    int a = idx % Ag, n = (idx / Ag) % Nn, h = idx / (Ag * Nn);
    int i0, i1; float f;
    lin_coord((n + 0.5f) * (197.0f / 3136.0f) - 0.5f, 197, i0, i1, f);
    float v0 = ab_val(na, hab, wab, cab, h, i0, a);
    float v1 = ab_val(na, hab, wab, cab, h, i1, a);
    bias2[idx] = (1.f - f) * v0 + f * v1;
}

// -------- stage 1 scores: scale*(agent . K^T) + bias1, 2x2 reg tiles ----------
__global__ __launch_bounds__(256) void k_scores1(const bf16* __restrict__ agentb,
                                                 const bf16* __restrict__ kb,
                                                 const float* __restrict__ bias1,
                                                 float* __restrict__ sf) {
    int lane = threadIdx.x & 31;
    int tile = blockIdx.x * (blockDim.x >> 5) + (threadIdx.x >> 5);
    int bh = tile / (2 * 98);
    int t  = tile % (2 * 98);
    int mt = (t / 98) * 32;
    int nt = (t % 98) * 32;
    int b = bh >> 3, h = bh & 7;
    v8f s00 = {}, s01 = {}, s10 = {}, s11 = {};
    const bf16* ar0 = agentb + ((size_t)(b * 64) + mt + (lane & 15)) * Cc + h * Dd;
    const bf16* ar1 = ar0 + (size_t)16 * Cc;
    const bf16* br0 = kb + ((size_t)(b * Nn) + nt + (lane & 15)) * Cc + h * Dd;
    const bf16* br1 = br0 + (size_t)16 * Cc;
#pragma unroll
    for (int k0 = 0; k0 < Dd; k0 += 32) {
        v16bf a0 = wmma_load_a(ar0 + k0, lane);
        v16bf a1 = wmma_load_a(ar1 + k0, lane);
        v16bf b0 = wmma_load_bt(br0 + k0, lane);
        v16bf b1 = wmma_load_bt(br1 + k0, lane);
        s00 = wmma_bf16(a0, b0, s00);
        s01 = wmma_bf16(a0, b1, s01);
        s10 = wmma_bf16(a1, b0, s10);
        s11 = wmma_bf16(a1, b1, s11);
    }
    int hi8 = (lane & 16) ? 8 : 0;
#pragma unroll
    for (int mo = 0; mo < 2; ++mo) {
#pragma unroll
        for (int jo = 0; jo < 2; ++jo) {
            v8f acc = mo ? (jo ? s11 : s10) : (jo ? s01 : s00);
            int n = nt + jo * 16 + (lane & 15);
            int mbase = mt + mo * 16 + hi8;
#pragma unroll
            for (int r = 0; r < 8; ++r) {
                int m = mbase + r;
                if (m < Ag)
                    sf[((size_t)bh * Ag + m) * Nn + n] =
                        acc[r] * 0.125f + bias1[((size_t)h * Ag + m) * Nn + n];
            }
        }
    }
}

// ---------------- stage 1 softmax over n (3136); write bf16 rows padded ---------
__global__ void k_softmax1(const float* __restrict__ sf, bf16* __restrict__ sb) {
    int row = blockIdx.x;            // bh*64 + a
    int bh = row >> 6, a = row & 63;
    bf16* orow = sb + (size_t)row * Nn;
    int t = threadIdx.x;
    if (a >= Ag) {
        for (int i = t; i < Nn; i += blockDim.x) orow[i] = (bf16)0.0f;
        return;
    }
    const float* irow = sf + ((size_t)bh * Ag + a) * Nn;
    __shared__ float red[256];
    float m = -1e30f;
    for (int i = t; i < Nn; i += blockDim.x) m = fmaxf(m, irow[i]);
    red[t] = m; __syncthreads();
    for (int s = 128; s > 0; s >>= 1) { if (t < s) red[t] = fmaxf(red[t], red[t + s]); __syncthreads(); }
    m = red[0]; __syncthreads();
    float sum = 0.f;
    for (int i = t; i < Nn; i += blockDim.x) sum += __expf(irow[i] - m);
    red[t] = sum; __syncthreads();
    for (int s = 128; s > 0; s >>= 1) { if (t < s) red[t] += red[t + s]; __syncthreads(); }
    float inv = 1.0f / red[0];
    for (int i = t; i < Nn; i += blockDim.x) orow[i] = (bf16)(__expf(irow[i] - m) * inv);
}

// -- agent_v = attn1(64p x 3136) . V(3136 x 64), 2x2 reg tiles; stores TRANSPOSED
__global__ __launch_bounds__(256) void k_agentv(const bf16* __restrict__ sb,
                                                const bf16* __restrict__ vtb,
                                                bf16* __restrict__ avt) {
    int lane = threadIdx.x & 31;
    int tile = blockIdx.x * (blockDim.x >> 5) + (threadIdx.x >> 5);
    int bh = tile >> 2;
    int t  = tile & 3;
    int mt = (t >> 1) * 32;
    int nt = (t & 1) * 32;
    int b = bh >> 3, h = bh & 7;
    v8f s00 = {}, s01 = {}, s10 = {}, s11 = {};
    const bf16* ar0 = sb + ((size_t)bh * 64 + mt + (lane & 15)) * Nn;
    const bf16* ar1 = ar0 + (size_t)16 * Nn;
    const bf16* br0 = vtb + ((size_t)(b * Cc) + h * Dd + nt + (lane & 15)) * Nn;
    const bf16* br1 = br0 + (size_t)16 * Nn;
#pragma unroll 2
    for (int k0 = 0; k0 < Nn; k0 += 32) {
        __builtin_prefetch(br0 + k0 + 128, 0, 1);
        v16bf a0 = wmma_load_a(ar0 + k0, lane);
        v16bf a1 = wmma_load_a(ar1 + k0, lane);
        v16bf b0 = wmma_load_bt(br0 + k0, lane);
        v16bf b1 = wmma_load_bt(br1 + k0, lane);
        s00 = wmma_bf16(a0, b0, s00);
        s01 = wmma_bf16(a0, b1, s01);
        s10 = wmma_bf16(a1, b0, s10);
        s11 = wmma_bf16(a1, b1, s11);
    }
    int hi8 = (lane & 16) ? 8 : 0;
#pragma unroll
    for (int mo = 0; mo < 2; ++mo) {
#pragma unroll
        for (int jo = 0; jo < 2; ++jo) {
            v8f acc = mo ? (jo ? s11 : s10) : (jo ? s01 : s00);
            int m = mt + mo * 16 + hi8;
            int n = nt + jo * 16 + (lane & 15);
#pragma unroll
            for (int r = 0; r < 8; ++r)
                avt[((size_t)bh * 64 + n) * 64 + m + r] = (bf16)acc[r];
        }
    }
}

// -------- stage 2 scores: scale*(Q . agent^T) + bias2, 2x2 reg tiles ----------
__global__ __launch_bounds__(256) void k_scores2(const bf16* __restrict__ qb,
                                                 const bf16* __restrict__ agentb,
                                                 const float* __restrict__ bias2,
                                                 float* __restrict__ sf) {
    int lane = threadIdx.x & 31;
    int tile = blockIdx.x * (blockDim.x >> 5) + (threadIdx.x >> 5);
    int bh = tile / (98 * 2);
    int t  = tile % (98 * 2);
    int mt = (t >> 1) * 32;
    int nt = (t & 1) * 32;
    int b = bh >> 3, h = bh & 7;
    v8f s00 = {}, s01 = {}, s10 = {}, s11 = {};
    const bf16* ar0 = qb + ((size_t)(b * Nn) + mt + (lane & 15)) * Cc + h * Dd;
    const bf16* ar1 = ar0 + (size_t)16 * Cc;
    const bf16* br0 = agentb + ((size_t)(b * 64) + nt + (lane & 15)) * Cc + h * Dd;
    const bf16* br1 = br0 + (size_t)16 * Cc;
#pragma unroll
    for (int k0 = 0; k0 < Dd; k0 += 32) {
        v16bf a0 = wmma_load_a(ar0 + k0, lane);
        v16bf a1 = wmma_load_a(ar1 + k0, lane);
        v16bf b0 = wmma_load_bt(br0 + k0, lane);
        v16bf b1 = wmma_load_bt(br1 + k0, lane);
        s00 = wmma_bf16(a0, b0, s00);
        s01 = wmma_bf16(a0, b1, s01);
        s10 = wmma_bf16(a1, b0, s10);
        s11 = wmma_bf16(a1, b1, s11);
    }
    int hi8 = (lane & 16) ? 8 : 0;
#pragma unroll
    for (int mo = 0; mo < 2; ++mo) {
#pragma unroll
        for (int jo = 0; jo < 2; ++jo) {
            v8f acc = mo ? (jo ? s11 : s10) : (jo ? s01 : s00);
            int acol = nt + jo * 16 + (lane & 15);
            if (acol < Ag) {
                int mbase = mt + mo * 16 + hi8;
#pragma unroll
                for (int r = 0; r < 8; ++r) {
                    int m = mbase + r;
                    sf[((size_t)bh * Nn + m) * Ag + acol] =
                        acc[r] * 0.125f + bias2[((size_t)h * Nn + m) * Ag + acol];
                }
            }
        }
    }
}

// ---------------- stage 2 softmax over a (49), one wave per row ---------------
__global__ void k_softmax2(const float* __restrict__ sf, bf16* __restrict__ sb) {
    int lane = threadIdx.x & 31;
    int row  = blockIdx.x * (blockDim.x >> 5) + (threadIdx.x >> 5); // bh*Nn + n
    const float* irow = sf + (size_t)row * Ag;
    float v0 = (lane < Ag)      ? irow[lane]      : -1e30f;
    float v1 = (lane + 32 < Ag) ? irow[lane + 32] : -1e30f;
    float m = fmaxf(v0, v1);
    for (int o = 16; o > 0; o >>= 1) m = fmaxf(m, __shfl_xor(m, o));
    float e0 = (lane < Ag)      ? __expf(v0 - m) : 0.f;
    float e1 = (lane + 32 < Ag) ? __expf(v1 - m) : 0.f;
    float s = e0 + e1;
    for (int o = 16; o > 0; o >>= 1) s += __shfl_xor(s, o);
    float inv = 1.0f / s;
    bf16* orow = sb + (size_t)row * 64;
    orow[lane]      = (bf16)(e0 * inv);
    orow[lane + 32] = (bf16)(e1 * inv);
}

// --- out = attn2 . agent_v  (+ fused depthwise-3x3 conv on v + b_dwc) ---------
__global__ __launch_bounds__(256) void k_out(const bf16* __restrict__ sb,
                                             const bf16* __restrict__ avt,
                                             const bf16* __restrict__ vb,
                                             const float* __restrict__ wdwc,
                                             const float* __restrict__ bdwc,
                                             float* __restrict__ outacc) {
    int lane = threadIdx.x & 31;
    int tile = blockIdx.x * (blockDim.x >> 5) + (threadIdx.x >> 5);
    int bh = tile / (196 * 4);
    int t  = tile % (196 * 4);
    int mt = (t >> 2) * 16;
    int nt = (t & 3) * 16;
    int b = bh >> 3, h = bh & 7;
    v8f acc = {};
    const bf16* arow = sb + ((size_t)bh * Nn + mt + (lane & 15)) * 64;
    const bf16* brow = avt + ((size_t)bh * 64 + nt + (lane & 15)) * 64;
    acc = wmma_bf16(wmma_load_a(arow, lane),      wmma_load_bt(brow, lane),      acc);
    acc = wmma_bf16(wmma_load_a(arow + 32, lane), wmma_load_bt(brow + 32, lane), acc);
    int m = mt + ((lane & 16) ? 8 : 0);
    int d = nt + (lane & 15);
    int c = h * Dd + d;
    float wc[9];
#pragma unroll
    for (int i = 0; i < 9; ++i) wc[i] = wdwc[c * 9 + i];
    float bterm = bdwc[c];
#pragma unroll
    for (int r = 0; r < 8; ++r) {
        int nrow = m + r;
        int y = nrow / W56, x = nrow % W56;
        float s = acc[r] + bterm;
#pragma unroll
        for (int ky = 0; ky < 3; ++ky) {
            int yy = y + ky - 1;
            if (yy < 0 || yy >= W56) continue;
#pragma unroll
            for (int kx = 0; kx < 3; ++kx) {
                int xx = x + kx - 1;
                if (xx < 0 || xx >= W56) continue;
                s += (float)vb[((size_t)(b * Nn) + yy * W56 + xx) * Cc + c] * wc[ky * 3 + kx];
            }
        }
        outacc[((size_t)(b * Nn) + nrow) * Cc + c] = s;
    }
}

// ---------------- final projection: (50176 x 512) * (512 x 512)^T + b ---------
__global__ __launch_bounds__(256) void k_proj(const bf16* __restrict__ ob,
                                              const bf16* __restrict__ wb,
                                              const float* __restrict__ bias,
                                              float* __restrict__ out) {
    int lane = threadIdx.x & 31;
    int tile = blockIdx.x * (blockDim.x >> 5) + (threadIdx.x >> 5);
    const int TN2 = Cc / 32; // 16
    int mt = (tile / TN2) * 32;
    int jt = (tile % TN2) * 32;
    v8f a00 = {}, a01 = {}, a10 = {}, a11 = {};
    const bf16* arow0 = ob + (size_t)(mt + (lane & 15)) * Cc;
    const bf16* arow1 = arow0 + (size_t)16 * Cc;
    const bf16* brow0 = wb + (size_t)(jt + (lane & 15)) * Cc;
    const bf16* brow1 = brow0 + (size_t)16 * Cc;
#pragma unroll 4
    for (int k0 = 0; k0 < Cc; k0 += 32) {
        v16bf a0 = wmma_load_a(arow0 + k0, lane);
        v16bf a1 = wmma_load_a(arow1 + k0, lane);
        v16bf b0 = wmma_load_bt(brow0 + k0, lane);
        v16bf b1 = wmma_load_bt(brow1 + k0, lane);
        a00 = wmma_bf16(a0, b0, a00);
        a01 = wmma_bf16(a0, b1, a01);
        a10 = wmma_bf16(a1, b0, a10);
        a11 = wmma_bf16(a1, b1, a11);
    }
    int hi8 = (lane & 16) ? 8 : 0;
#pragma unroll
    for (int mo = 0; mo < 2; ++mo) {
#pragma unroll
        for (int jo = 0; jo < 2; ++jo) {
            v8f acc = mo ? (jo ? a11 : a10) : (jo ? a01 : a00);
            int j = jt + jo * 16 + (lane & 15);
            float bj = bias[j];
            int mrow = mt + mo * 16 + hi8;
#pragma unroll
            for (int r = 0; r < 8; ++r) out[(size_t)(mrow + r) * Cc + j] = acc[r] + bj;
        }
    }
}

// ---------------- host orchestration ----------------
extern "C" void kernel_launch(void* const* d_in, const int* in_sizes, int n_in,
                              void* d_out, int out_size, void* d_ws, size_t ws_size,
                              hipStream_t stream) {
    const float* x       = (const float*)d_in[0];
    const float* w_qkv   = (const float*)d_in[1];
    const float* w_proj  = (const float*)d_in[2];
    const float* b_proj  = (const float*)d_in[3];
    const float* w_dwc   = (const float*)d_in[4];
    const float* b_dwc   = (const float*)d_in[5];
    const float* an_bias = (const float*)d_in[6];
    const float* na_bias = (const float*)d_in[7];
    const float* ah_bias = (const float*)d_in[8];
    const float* aw_bias = (const float*)d_in[9];
    const float* ha_bias = (const float*)d_in[10];
    const float* wa_bias = (const float*)d_in[11];
    const float* ca_bias = (const float*)d_in[12];
    float* out = (float*)d_out;

    // workspace layout (256B aligned slabs)
    char* w = (char*)d_ws;
    size_t o = 0;
    auto nb = [](size_t x) { return (x + 255) & ~(size_t)255; };
    bf16*  xb      = (bf16*)(w + o); o += nb((size_t)Bb * Nn * Cc * 2);
    bf16*  wqkvb   = (bf16*)(w + o); o += nb((size_t)3 * Cc * Cc * 2);
    bf16*  wprojb  = (bf16*)(w + o); o += nb((size_t)Cc * Cc * 2);
    bf16*  qb      = (bf16*)(w + o); o += nb((size_t)Bb * Nn * Cc * 2);
    bf16*  kb      = (bf16*)(w + o); o += nb((size_t)Bb * Nn * Cc * 2);
    bf16*  vb      = (bf16*)(w + o); o += nb((size_t)Bb * Nn * Cc * 2);
    bf16*  vtb     = (bf16*)(w + o); o += nb((size_t)Bb * Nn * Cc * 2);
    bf16*  agentb  = (bf16*)(w + o); o += nb((size_t)Bb * 64 * Cc * 2);
    bf16*  avt     = (bf16*)(w + o); o += nb((size_t)Bb * Hh * 64 * 64 * 2);
    float* bias1   = (float*)(w + o); o += nb((size_t)Hh * Ag * Nn * 4);
    float* bias2   = (float*)(w + o); o += nb((size_t)Hh * Nn * Ag * 4);
    float* sf      = (float*)(w + o); o += nb((size_t)Bb * Hh * Ag * Nn * 4); // shared s1/s2
    bf16*  sb      = (bf16*)(w + o); o += nb((size_t)Bb * Hh * 64 * Nn * 2);  // shared s1/s2
    float* outacc  = (float*)(w + o); o += nb((size_t)Bb * Nn * Cc * 4);
    bf16*  outb    = xb; // xb is dead after the qkv GEMM

    const int TB = 256;
    // 1) casts to bf16 (8 elements per thread, vectorized)
    {
        int n8 = (Bb * Nn * Cc) / 8;
        k_cast8<<<(n8 + TB - 1) / TB, TB, 0, stream>>>(x, xb, n8);
        n8 = (3 * Cc * Cc) / 8;
        k_cast8<<<(n8 + TB - 1) / TB, TB, 0, stream>>>(w_qkv, wqkvb, n8);
        n8 = (Cc * Cc) / 8;
        k_cast8<<<(n8 + TB - 1) / TB, TB, 0, stream>>>(w_proj, wprojb, n8);
    }
    // 2) qkv GEMM: 32x32 tiles, 8 waves per block
    k_qkv<<<((Bb * Nn / 32) * (3 * Cc / 32)) / 8, TB, 0, stream>>>(xb, wqkvb, qb, kb, vb);
    // 3) V transpose (async global->LDS) + agent pooling
    k_vt<<<Bb * (Cc / 32) * (Nn / 32), dim3(32, 8), 0, stream>>>(vb, vtb);
    k_pool<<<(Bb * 64 * Cc) / TB, TB, 0, stream>>>(qb, agentb);
    // 4) bias tables
    {
        int n = Hh * Ag * Nn;
        k_bias1<<<(n + TB - 1) / TB, TB, 0, stream>>>(an_bias, ah_bias, aw_bias, bias1);
        n = Hh * Nn * Ag;
        k_bias2<<<(n + TB - 1) / TB, TB, 0, stream>>>(na_bias, ha_bias, wa_bias, ca_bias, bias2);
    }
    // 5) stage-1 scores + softmax + agent_v
    k_scores1<<<(Bb * Hh * 2 * 98) / 8, TB, 0, stream>>>(agentb, kb, bias1, sf);
    k_softmax1<<<Bb * Hh * 64, TB, 0, stream>>>(sf, sb);
    k_agentv<<<(Bb * Hh * 4) / 8, TB, 0, stream>>>(sb, vtb, avt);
    // 6) stage-2 scores + softmax + output(+fused dwc)
    k_scores2<<<(Bb * Hh * 98 * 2) / 8, TB, 0, stream>>>(qb, agentb, bias2, sf);
    k_softmax2<<<(Bb * Hh * Nn) / 8, TB, 0, stream>>>(sf, sb);
    k_out<<<(Bb * Hh * 196 * 4) / 8, TB, 0, stream>>>(sb, avt, vb, w_dwc, b_dwc, outacc);
    // 7) cast accumulated output and project
    {
        int n8 = (Bb * Nn * Cc) / 8;
        k_cast8<<<(n8 + TB - 1) / TB, TB, 0, stream>>>(outacc, outb, n8);
    }
    k_proj<<<((Bb * Nn / 32) * (Cc / 32)) / 8, TB, 0, stream>>>(outb, wprojb, b_proj, out);

    (void)in_sizes; (void)n_in; (void)out_size; (void)ws_size;
}
